// Block_41695542510261
// MI455X (gfx1250) — matrix operand order repaired and checked by reading, not compiled, not run on hardware
//
#include <hip/hip_runtime.h>

typedef __bf16 bf16_t;
typedef bf16_t v16bf __attribute__((ext_vector_type(16)));
typedef float  v8f   __attribute__((ext_vector_type(8)));

#define DEVI __device__ __forceinline__

// ---- CDNA5 async global->LDS copy (probe-verified on this toolchain) ------
#if defined(__has_builtin)
#  if __has_builtin(__builtin_amdgcn_global_load_async_to_lds_b128) && \
      __has_builtin(__builtin_amdgcn_s_wait_asynccnt)
#    define USE_ASYNC_LDS 1
#  endif
#endif
#ifndef USE_ASYNC_LDS
#  define USE_ASYNC_LDS 0
#endif

// Builtin takes pointers to 16-byte int vectors in AS1 (global) / AS3 (LDS).
typedef int v4i_vec __attribute__((vector_size(16)));
typedef __attribute__((address_space(1))) v4i_vec* g_v4i_ptr;
typedef __attribute__((address_space(3))) v4i_vec* l_v4i_ptr;

namespace {

constexpr int kB  = 4;
constexpr int kT  = 1024;
constexpr int kD  = 1024;
constexpr int kH  = 16;
constexpr int kDh = 64;
constexpr int kDff = 4096;
constexpr int kM  = kB * kT;       // 4096 token rows
constexpr int kQKVld = 3 * kD;     // 3072
constexpr int kBPad = 72;          // LDS tile row stride (elements)

DEVI unsigned short f2bf(float f) {
  unsigned int u = __builtin_bit_cast(unsigned int, f);
  u += 0x7FFFu + ((u >> 16) & 1u);          // round-to-nearest-even
  return (unsigned short)(u >> 16);
}
DEVI float bf2f(unsigned short s) {
  unsigned int u = ((unsigned int)s) << 16;
  return __builtin_bit_cast(float, u);
}

union FragAB {
  v16bf          v;
  unsigned short h[16];
  unsigned int   w[8];
};

DEVI v8f vzero() { v8f z = {0.f,0.f,0.f,0.f,0.f,0.f,0.f,0.f}; return z; }

DEVI v8f wmma_bf16(const FragAB& a, const FragAB& b, v8f c) {
  return __builtin_amdgcn_wmma_f32_16x16x32_bf16(false, a.v, false, b.v,
                                                 (short)0, c, false, false);
}

// 16-byte global -> LDS copy: async DMA on CDNA5, sync fallback otherwise.
DEVI void stage16(const unsigned short* g, unsigned short* l) {
#if USE_ASYNC_LDS
  __builtin_amdgcn_global_load_async_to_lds_b128(
      (g_v4i_ptr)g, (l_v4i_ptr)l, 0, 0);
#else
  *reinterpret_cast<uint4*>(l) = *reinterpret_cast<const uint4*>(g);
#endif
}
DEVI void stage_wait0() {
#if USE_ASYNC_LDS
  __builtin_amdgcn_s_wait_asynccnt(0);
#endif
}

// A fragment: 16x32 bf16 tile from a row-major matrix (ld in elements).
// ISA layout: lane 0-15 -> M rows (K 0..15 side), lanes 16-31 same rows K-shifted;
// dword d holds K pair {base+2(d&3), base+2(d&3)+1}, base = (d>=4?16:0) + (laneHi?8:0).
DEVI void load_a_frag(FragAB& f, const unsigned short* __restrict__ base,
                      int ld, int lane) {
  const int m  = lane & 15;
  const int hb = (lane >> 4) ? 8 : 0;
  const unsigned short* row = base + (size_t)m * ld;
#pragma unroll
  for (int d = 0; d < 8; ++d) {
    const int k = ((d & 4) ? 16 : 0) + hb + (d & 3) * 2;
    f.w[d] = *reinterpret_cast<const unsigned int*>(row + k);
  }
}

// B fragment from LDS tile (row-major, stride kBPad): lane%16 = N column,
// element s = K row in table order (same as A).
DEVI void load_b_frag_lds(FragAB& f, const unsigned short* col, int lane) {
  const int hb = (lane >> 4) ? 8 : 0;
#pragma unroll
  for (int s = 0; s < 16; ++s)
    f.h[s] = col[(((s & 8) ? 16 : 0) + hb + (s & 7)) * kBPad];
}

// Butterfly reductions within each 16-lane half of the wave32
// (a C-tile row's 16 columns live in one half).
DEVI float halfmax(float x) {
#pragma unroll
  for (int off = 1; off <= 8; off <<= 1) x = fmaxf(x, __shfl_xor(x, off, 32));
  return x;
}
DEVI float halfsum(float x) {
#pragma unroll
  for (int off = 1; off <= 8; off <<= 1) x += __shfl_xor(x, off, 32);
  return x;
}

// ---------------------------------------------------------------- conversions

__global__ void k_cvt_head(const float* __restrict__ w,
                           unsigned short* __restrict__ dst, int colOff) {
  // w: [16][1024][64]  ->  dst[c][colOff + h*64 + d], dst ld = 3072
  const size_t n = (size_t)kH * kD * kDh;   // 1,048,576
  size_t i = (size_t)blockIdx.x * blockDim.x + threadIdx.x;
  if (i >= n) return;
  int d = (int)(i & 63);
  size_t r = i >> 6;
  int c  = (int)(r & 1023);
  int hh = (int)(r >> 10);
  dst[(size_t)c * kQKVld + colOff + hh * kDh + d] = f2bf(w[i]);
}

__global__ void k_cvt_mat(const float* __restrict__ w,
                          unsigned short* __restrict__ dst,
                          long long n, int cols, int ldDst, int colOff) {
  size_t i = (size_t)blockIdx.x * blockDim.x + threadIdx.x;
  if (i >= (size_t)n) return;
  int c = (int)(i % cols);
  size_t r = i / cols;
  dst[r * ldDst + colOff + c] = f2bf(w[i]);
}

__global__ void k_concat_bias(const float* __restrict__ a,
                              const float* __restrict__ b,
                              float* __restrict__ out, int n1, int n2) {
  int i = blockIdx.x * blockDim.x + threadIdx.x;
  if (i < n1) out[i] = a[i];
  else if (i < n1 + n2) out[i] = b[i - n1];
}

// ---------------------------------------------------------------- layer norm

__global__ void k_layernorm(const float* __restrict__ x,
                            const float* __restrict__ g,
                            const float* __restrict__ be,
                            unsigned short* __restrict__ out, int C) {
  __shared__ float sred[256];
  const int row = blockIdx.x, tid = threadIdx.x;
  const float* xr = x + (size_t)row * C;
  float s = 0.f;
  for (int i = tid; i < C; i += 256) s += xr[i];
  sred[tid] = s; __syncthreads();
  for (int st = 128; st > 0; st >>= 1) {
    if (tid < st) sred[tid] += sred[tid + st];
    __syncthreads();
  }
  const float mu = sred[0] / C;
  __syncthreads();
  float vs = 0.f;
  for (int i = tid; i < C; i += 256) { float d = xr[i] - mu; vs += d * d; }
  sred[tid] = vs; __syncthreads();
  for (int st = 128; st > 0; st >>= 1) {
    if (tid < st) sred[tid] += sred[tid + st];
    __syncthreads();
  }
  const float inv = rsqrtf(sred[0] / C + 1e-5f);
  unsigned short* orow = out + (size_t)row * C;
  for (int i = tid; i < C; i += 256)
    orow[i] = f2bf((xr[i] - mu) * inv * g[i] + be[i]);
}

// ---------------------------------------------------------------- WMMA GEMM
// C[M,N] = A[M,K](bf16) * B[K,N](bf16) [+ bias[col]] [+ res[row,col]]
// grid = (N/64, M/128), block = 256 (8 waves); wave w owns rows w*16..+16.
// The 32x64 B tile (shared by all 8 waves) is double-buffered in LDS via
// async global->LDS DMA: wait own ASYNCcnt, barrier, kick next tile's DMA,
// then load all 4 B fragments from LDS and run the 4-wmma chain.

__global__ __launch_bounds__(256)
void k_gemm(const unsigned short* __restrict__ A,
            const unsigned short* __restrict__ Bm,
            int K, int lda, int ldb, int ldc,
            const float* __restrict__ bias,
            const float* __restrict__ res, int ldres,
            unsigned short* __restrict__ outb,
            float* __restrict__ outf) {
  __shared__ unsigned short Bls[2][32 * kBPad];

  const int lane = threadIdx.x & 31;
  const int wave = threadIdx.x >> 5;
  const int rowBase = blockIdx.y * 128 + wave * 16;
  const int colBase = blockIdx.x * 64;

  // B staging: 256 threads x 16B == one 32(K) x 64(N) bf16 tile.
  const int scr = threadIdx.x >> 3;        // k-row in tile: 0..31
  const int scc = (threadIdx.x & 7) * 8;   // col in tile:   0,8,..,56

  v8f acc[4];
#pragma unroll
  for (int t = 0; t < 4; ++t) acc[t] = vzero();

  const unsigned short* srcB = Bm + (size_t)scr * ldb + colBase + scc;
  const int ldsOff = scr * kBPad + scc;

  stage16(srcB, &Bls[0][0] + ldsOff);      // prologue: k=0 -> buffer 0

  for (int k = 0; k < K; k += 32) {
    const int cur = (k >> 5) & 1;
    stage_wait0();                         // own DMA for buffer `cur` done
    __syncthreads();                       // all waves staged + prior reads done
    if (k + 32 < K)
      stage16(srcB + (size_t)(k + 32) * ldb, &Bls[cur ^ 1][0] + ldsOff);

    FragAB a, bfr[4];
    load_a_frag(a, A + (size_t)rowBase * lda + k, lda, lane);
#pragma unroll
    for (int t = 0; t < 4; ++t)            // issue all LDS loads up front
      load_b_frag_lds(bfr[t], &Bls[cur][t * 16 + (lane & 15)], lane);
#pragma unroll
    for (int t = 0; t < 4; ++t)
      acc[t] = wmma_bf16(a, bfr[t], acc[t]);
  }

  const int hi = lane >> 4, nloc = lane & 15;
#pragma unroll
  for (int t = 0; t < 4; ++t) {
#pragma unroll
    for (int v = 0; v < 8; ++v) {
      const int r = rowBase + v + 8 * hi;
      const int c = colBase + t * 16 + nloc;
      float val = acc[t][v];
      if (bias) val += bias[c];
      if (res)  val += res[(size_t)r * ldres + c];
      if (outb) outb[(size_t)r * ldc + c] = f2bf(val);
      else      outf[(size_t)r * ldc + c] = val;
    }
  }
}

// ------------------------------------------------------- flash attention
// qkv: [B*T, 3072] bf16 (q|k|v each 1024 wide, head h at h*64).
// Block = 4 waves = 64 query rows of one (b,h); block-uniform causal loop
// over 32-key blocks with K/V tiles double-buffered in LDS via async DMA.
// Each wave owns 16 query rows; waves past their causal bound skip compute
// but keep staging/barriers uniform. ctx out: [B*T, 1024] bf16.

__global__ __launch_bounds__(128)
void k_flash(const unsigned short* __restrict__ qkv,
             unsigned short* __restrict__ ctx) {
  __shared__ unsigned short Kls[2][32 * kBPad];
  __shared__ unsigned short Vls[2][32 * kBPad];
  __shared__ unsigned short Pls[4][16 * 36];   // per-wave P staging, padded

  const int tid  = threadIdx.x;
  const int lane = tid & 31;
  const int wave = tid >> 5;
  const int hi   = lane >> 4, nloc = lane & 15;
  const int h = blockIdx.y, b = blockIdx.z;
  const int rowBlk = blockIdx.x * 64;
  const int qRow0  = rowBlk + wave * 16;

  const unsigned short* base = qkv + (size_t)b * kT * kQKVld;

  FragAB aq[2];                                 // Q 16x64 as two K=32 A-frags
#pragma unroll
  for (int c = 0; c < 2; ++c)
    load_a_frag(aq[c], base + (size_t)qRow0 * kQKVld + h * kDh + c * 32,
                kQKVld, lane);

  v8f acc[4];
  float m_run[8], l_run[8];
#pragma unroll
  for (int t = 0; t < 4; ++t) acc[t] = vzero();
#pragma unroll
  for (int v = 0; v < 8; ++v) { m_run[v] = -1e30f; l_run[v] = 0.f; }

  const float scl = 0.03125f;                   // 1/sqrt(1024)
  const int nkb    = (qRow0 + 47) >> 5;         // this wave's causal bound
  const int nkbMax = (rowBlk + 95) >> 5;        // block bound (wave 3)

  // K/V staging: 128 threads x 2 chunks x 16B per tile (32 rows x 64 cols).
  auto stageKV = [&](int buf, int kb) {
#pragma unroll
    for (int cc = 0; cc < 2; ++cc) {
      const int ch  = tid + cc * 128;           // chunk 0..255
      const int r   = ch >> 3;                  // key row in tile
      const int col = (ch & 7) * 8;             // col in tile
      const size_t grow = (size_t)(kb * 32 + r) * kQKVld;
      stage16(base + grow + kD     + h * kDh + col, &Kls[buf][0] + r * kBPad + col);
      stage16(base + grow + 2 * kD + h * kDh + col, &Vls[buf][0] + r * kBPad + col);
    }
  };

  stageKV(0, 0);                                // prologue

  for (int kb = 0; kb < nkbMax; ++kb) {
    const int cur = kb & 1;
    stage_wait0();                              // own K/V DMAs for `cur` done
    __syncthreads();                            // whole tile staged everywhere
    if (kb + 1 < nkbMax) stageKV(cur ^ 1, kb + 1);

    if (kb < nkb) {
      // ---- S = Q K^T : B[d][n] = Kls[j*16+n][c*32 + d]  (LDS gather)
      FragAB bk[2][2];
#pragma unroll
      for (int j = 0; j < 2; ++j) {
#pragma unroll
        for (int c = 0; c < 2; ++c) {
#pragma unroll
          for (int n = 0; n < 16; ++n)
            bk[j][c].h[n] = Kls[cur][(j * 16 + nloc) * kBPad + c * 32 + ((n & 8) ? 16 : 0) + (hi ? 8 : 0) + (n & 7)];
        }
      }
      v8f s0 = vzero(), s1 = vzero();
      s0 = wmma_bf16(aq[0], bk[0][0], s0);
      s0 = wmma_bf16(aq[1], bk[0][1], s0);
      s1 = wmma_bf16(aq[0], bk[1][0], s1);
      s1 = wmma_bf16(aq[1], bk[1][1], s1);

      // ---- scale + causal mask + online softmax
      float rm[8], alpha[8];
#pragma unroll
      for (int v = 0; v < 8; ++v) {
        const int mrow = qRow0 + v + 8 * hi;
        float a0 = s0[v] * scl;
        float a1 = s1[v] * scl;
        if (kb * 32 + nloc      > mrow) a0 = -1e30f;
        if (kb * 32 + 16 + nloc > mrow) a1 = -1e30f;
        s0[v] = a0; s1[v] = a1;
        rm[v] = halfmax(fmaxf(a0, a1));
      }
#pragma unroll
      for (int v = 0; v < 8; ++v) {
        const float nm = fmaxf(m_run[v], rm[v]);
        const float p0 = __expf(s0[v] - nm);
        const float p1 = __expf(s1[v] - nm);
        const float rs = halfsum(p0 + p1);
        const float al = __expf(m_run[v] - nm);
        l_run[v] = l_run[v] * al + rs;
        m_run[v] = nm;
        s0[v] = p0; s1[v] = p1;
        alpha[v] = al;
      }
#pragma unroll
      for (int t = 0; t < 4; ++t)
#pragma unroll
        for (int v = 0; v < 8; ++v) acc[t][v] *= alpha[v];

      // ---- C-layout P -> A-layout via per-wave LDS
#pragma unroll
      for (int v = 0; v < 8; ++v) {
        const int row = v + 8 * hi;
        Pls[wave][row * 36 + nloc]      = f2bf(s0[v]);
        Pls[wave][row * 36 + 16 + nloc] = f2bf(s1[v]);
      }
      __builtin_amdgcn_wave_barrier();
      FragAB pa;
      {
        const int hb = hi ? 8 : 0;
#pragma unroll
        for (int d = 0; d < 8; ++d) {
          const int k = ((d & 4) ? 16 : 0) + hb + (d & 3) * 2;
          pa.w[d] = *reinterpret_cast<const unsigned int*>(
              &Pls[wave][nloc * 36 + k]);
        }
      }

      // ---- O += P V : B[key][n=d] = Vls[key][t*16 + n] (contiguous LDS)
      FragAB vb[4];
#pragma unroll
      for (int t = 0; t < 4; ++t)
        load_b_frag_lds(vb[t], &Vls[cur][t * 16 + nloc], lane);
#pragma unroll
      for (int t = 0; t < 4; ++t)
        acc[t] = wmma_bf16(pa, vb[t], acc[t]);
    }
  }

  // ---- normalize + write ctx[b, t, h*64 + d]
#pragma unroll
  for (int v = 0; v < 8; ++v) {
    const float inv = 1.f / l_run[v];
    const int row = qRow0 + v + 8 * hi;
#pragma unroll
    for (int t = 0; t < 4; ++t)
      ctx[(size_t)(b * kT + row) * kD + h * kDh + t * 16 + nloc] =
          f2bf(acc[t][v] * inv);
  }
}

// ---------------------------------------------------------------- SwiGLU

__global__ void k_silu_mul(const unsigned short* __restrict__ gv,
                           unsigned short* __restrict__ u) {
  const size_t n = (size_t)kM * kDff;
  size_t i = (size_t)blockIdx.x * blockDim.x + threadIdx.x;
  if (i >= n) return;
  const size_t r = i >> 12;            // /4096
  const int    c = (int)(i & 4095);
  const float g = bf2f(gv[r * 8192 + c]);
  const float w = bf2f(gv[r * 8192 + 4096 + c]);
  const float s = g / (1.f + __expf(-g));
  u[i] = f2bf(s * w);
}

}  // namespace

// =================================================================== launch

extern "C" void kernel_launch(void* const* d_in, const int* in_sizes, int n_in,
                              void* d_out, int out_size, void* d_ws,
                              size_t ws_size, hipStream_t stream) {
  (void)in_sizes; (void)n_in; (void)out_size; (void)ws_size;

  const float* x   = (const float*)d_in[0];
  // d_in[1] (mask) is the causal -inf mask; implemented analytically.
  const float* Wq  = (const float*)d_in[2];
  const float* Wk  = (const float*)d_in[3];
  const float* Wv  = (const float*)d_in[4];
  const float* Wo  = (const float*)d_in[5];
  const float* bo  = (const float*)d_in[6];
  const float* w1  = (const float*)d_in[7];
  const float* b1  = (const float*)d_in[8];
  const float* w2  = (const float*)d_in[9];
  const float* b2  = (const float*)d_in[10];
  const float* w3  = (const float*)d_in[11];
  const float* b3  = (const float*)d_in[12];
  const float* g1  = (const float*)d_in[13];
  const float* be1 = (const float*)d_in[14];
  const float* g2  = (const float*)d_in[15];
  const float* be2 = (const float*)d_in[16];
  float* out = (float*)d_out;

  char* wsb = (char*)d_ws;
  size_t off = 0;
  auto alloc = [&](size_t bytes) -> char* {
    char* p = wsb + off;
    off += (bytes + 255) & ~(size_t)255;
    return p;
  };
  unsigned short* wqkv   = (unsigned short*)alloc((size_t)kD * kQKVld * 2);
  unsigned short* wo_b   = (unsigned short*)alloc((size_t)kD * kD * 2);
  unsigned short* w12_b  = (unsigned short*)alloc((size_t)kD * 2 * kDff * 2);
  unsigned short* w3_b   = (unsigned short*)alloc((size_t)kDff * kD * 2);
  float*          bias12 = (float*)alloc((size_t)2 * kDff * 4);
  unsigned short* h1     = (unsigned short*)alloc((size_t)kM * kD * 2);
  unsigned short* qkv    = (unsigned short*)alloc((size_t)kM * kQKVld * 2);
  unsigned short* ctx    = (unsigned short*)alloc((size_t)kM * kD * 2);
  float*          x1     = (float*)alloc((size_t)kM * kD * 4);
  unsigned short* h2     = (unsigned short*)alloc((size_t)kM * kD * 2);
  unsigned short* gv     = (unsigned short*)alloc((size_t)kM * 2 * kDff * 2);
  unsigned short* ub     = (unsigned short*)alloc((size_t)kM * kDff * 2);

  // ---- weight / bias conversion (fp32 -> bf16, with re-layout)
  {
    const int n = kH * kD * kDh;                    // 1,048,576
    dim3 g((n + 255) / 256);
    k_cvt_head<<<g, 256, 0, stream>>>(Wq, wqkv, 0);
    k_cvt_head<<<g, 256, 0, stream>>>(Wk, wqkv, kD);
    k_cvt_head<<<g, 256, 0, stream>>>(Wv, wqkv, 2 * kD);
  }
  k_cvt_mat<<<dim3((kD * kD + 255) / 256), 256, 0, stream>>>(
      Wo, wo_b, (long long)kD * kD, kD, kD, 0);
  k_cvt_mat<<<dim3((kD * kDff + 255) / 256), 256, 0, stream>>>(
      w1, w12_b, (long long)kD * kDff, kDff, 2 * kDff, 0);
  k_cvt_mat<<<dim3((kD * kDff + 255) / 256), 256, 0, stream>>>(
      w2, w12_b, (long long)kD * kDff, kDff, 2 * kDff, kDff);
  k_cvt_mat<<<dim3((kDff * kD + 255) / 256), 256, 0, stream>>>(
      w3, w3_b, (long long)kDff * kD, kD, kD, 0);
  k_concat_bias<<<dim3((2 * kDff + 255) / 256), 256, 0, stream>>>(
      b1, b2, bias12, kDff, kDff);

  // ---- h1 = LN(x; g1, be1)  (bf16)
  k_layernorm<<<dim3(kM), 256, 0, stream>>>(x, g1, be1, h1, kD);

  // ---- qkv = h1 @ [Wq|Wk|Wv]   (bf16 out, [4096, 3072])
  k_gemm<<<dim3(kQKVld / 64, kM / 128), 256, 0, stream>>>(
      h1, wqkv, kD, kD, kQKVld, kQKVld,
      nullptr, nullptr, 0, qkv, nullptr);

  // ---- ctx = causal flash attention (bf16, [4096, 1024], heads concat)
  k_flash<<<dim3(kT / 64, kH, kB), 128, 0, stream>>>(qkv, ctx);

  // ---- x1 = x + ctx @ Wo + bo   (fp32)
  k_gemm<<<dim3(kD / 64, kM / 128), 256, 0, stream>>>(
      ctx, wo_b, kD, kD, kD, kD,
      bo, x, kD, nullptr, x1);

  // ---- h2 = LN(x1; g2, be2) (bf16)
  k_layernorm<<<dim3(kM), 256, 0, stream>>>(x1, g2, be2, h2, kD);

  // ---- gv = h2 @ [w1|w2] + [b1|b2]   (bf16, [4096, 8192])
  k_gemm<<<dim3(2 * kDff / 64, kM / 128), 256, 0, stream>>>(
      h2, w12_b, kD, kD, 2 * kDff, 2 * kDff,
      bias12, nullptr, 0, gv, nullptr);

  // ---- u = silu(gate) * val   (bf16, [4096, 4096])
  k_silu_mul<<<dim3(((size_t)kM * kDff + 255) / 256), 256, 0, stream>>>(gv, ub);

  // ---- out = x1 + u @ w3 + b3   (fp32)
  k_gemm<<<dim3(kD / 64, kM / 128), 256, 0, stream>>>(
      ub, w3_b, kDff, kDff, kD, kD,
      b3, x1, kD, nullptr, out);
}
